// AttentionHead_Hybrid2_8426725834768
// MI455X (gfx1250) — matrix operand order repaired and checked by reading, not compiled, not run on hardware
//
#include <hip/hip_runtime.h>
#include <hip/hip_bf16.h>

typedef __attribute__((ext_vector_type(16))) _Float16 v16h;
typedef __attribute__((ext_vector_type(8)))  float    v8f;
typedef __attribute__((__vector_size__(16))) int      i32x4;   // int4, matches builtin param

#define BATCH 4
#define NTOK  4096
#define DK    64

#if defined(__gfx1250__) && __has_builtin(__builtin_amdgcn_global_load_async_to_lds_b128) && \
    __has_builtin(__builtin_amdgcn_s_wait_asynccnt)
#define USE_ASYNC_LDS 1
#else
#define USE_ASYNC_LDS 0
#endif

// ---------------------------------------------------------------------------
// Kernel 1: V = x @ Wv^T + bv (written as f16 WMMA B-fragments), Q = x.wq, K = x.wk
// Fragment layout per 32-row j-tile, per 16-col n-tile:
//   frag[lane*16 + t] = V[j0 + (lane>=16?16:0) + t][nt*16 + lane%16]   (f16)
// B operand (32x16) of v_wmma_f32_16x16x32_f16 in contiguous k order, same order as the A fragment.
// ---------------------------------------------------------------------------
__global__ void precompute_kernel(const float* __restrict__ x,
                                  const float* __restrict__ Wv,
                                  const float* __restrict__ bv,
                                  const float* __restrict__ wq,
                                  const float* __restrict__ wk,
                                  _Float16* __restrict__ Vfrag,
                                  float* __restrict__ Qb,
                                  float* __restrict__ Kb) {
    __shared__ float xs[32][65];                  // +1 pad: conflict-free column reads
    const int b   = blockIdx.x >> 7;              // NTOK/32 = 128 tiles per batch
    const int jt  = blockIdx.x & 127;
    const int j0  = jt * 32;
    const int tid = threadIdx.x;                  // 256

    #pragma unroll
    for (int k = 0; k < 8; ++k) {
        int e = tid + k * 256;
        xs[e >> 6][e & 63] = x[((size_t)(b * NTOK + j0) << 6) + e];
    }
    __syncthreads();

    if (tid < 64) {
        int r = tid >> 1;
        const float* wvec = (tid & 1) ? wk : wq;
        float acc = 0.f;
        #pragma unroll
        for (int d = 0; d < 64; ++d) acc += xs[r][d] * wvec[d];
        if (tid & 1) Kb[b * NTOK + j0 + r] = acc;
        else         Qb[b * NTOK + j0 + r] = acc;
    }

    #pragma unroll
    for (int k = 0; k < 8; ++k) {
        int e    = tid + k * 256;
        int nt   = e >> 9;
        int lane = (e >> 4) & 31;
        int t    = e & 15;
        int jl   = ((lane >> 4) << 4) + t;
        int n    = nt * 16 + (lane & 15);
        float acc = bv[n];
        const float* wrow = Wv + n * 64;
        #pragma unroll
        for (int d = 0; d < 64; ++d) acc += xs[jl][d] * wrow[d];
        Vfrag[(size_t)(b * 128 + jt) * 2048 + e] = (_Float16)acc;
    }
}

#if USE_ASYNC_LDS
// Wave 0 copies one 4 KB fragment tile global -> LDS (8 x b128 per lane, ASYNCcnt).
__device__ __forceinline__ void async_copy_tile(const _Float16* gsrc, _Float16* ldst, int lane) {
    const char* g = (const char*)gsrc + lane * 16;
    char*       l = (char*)ldst + lane * 16;
    #pragma unroll
    for (int i = 0; i < 8; ++i)
        __builtin_amdgcn_global_load_async_to_lds_b128((i32x4*)(g + i * 512),
                                                       (i32x4*)(l + i * 512), 0, 0);
}
#endif

// ---------------------------------------------------------------------------
// Kernel 2: fused RBF-score softmax-GEMM:  PV = softmax(exp(-(Q_i-K_j)^2)/8) @ V
// Scores in (0, 0.125] -> no max subtraction; second exp replaced by cubic
// Taylor poly (rel err ~1e-5, below f16 quantization).
// Grid: BATCH * (NTOK/64) blocks, 128 threads (4 waves, one 16-row i-tile each).
// ---------------------------------------------------------------------------
__global__ void attn_kernel(const _Float16* __restrict__ Vfrag,
                            const float* __restrict__ Qb,
                            const float* __restrict__ Kb,
                            float* __restrict__ PV) {
    __shared__ float Ks[NTOK];                    // 16 KB, shared by 4 waves
#if USE_ASYNC_LDS
    __shared__ __align__(16) _Float16 vtile[2][2048];  // double-buffered B tile (2 x 4 KB)
#endif
    const int b   = blockIdx.x >> 6;
    const int it  = blockIdx.x & 63;
    const int tid = threadIdx.x;                  // 128
    const int wave = tid >> 5;
    const int lane = tid & 31;

    #pragma unroll
    for (int k = 0; k < 32; ++k)
        Ks[tid + k * 128] = Kb[b * NTOK + tid + k * 128];

    const _Float16* vbase = Vfrag + (size_t)b * 128 * 2048;
#if USE_ASYNC_LDS
    if (wave == 0) async_copy_tile(vbase, &vtile[0][0], lane);   // prefetch tile 0
#endif
    __syncthreads();                              // Ks ready

    const int i0 = it * 64 + wave * 16;
    const int M  = lane & 15;                     // A-matrix row handled by this lane
    const int hb = (lane >> 4) << 3;              // lanes>=16 own K-range +16 (2*hb)
    const float q = Qb[(size_t)b * NTOK + i0 + M];

    v8f acc0 = {}, acc1 = {}, acc2 = {}, acc3 = {};
    float rsp[4] = {0.f, 0.f, 0.f, 0.f};          // partial row sums (short dep chains)

    for (int jt = 0; jt < 128; ++jt) {
        const int j0 = jt * 32;

#if USE_ASYNC_LDS
        const int cur = jt & 1;
        if (wave == 0) __builtin_amdgcn_s_wait_asynccnt(0);  // tile jt landed in LDS
        __syncthreads();                                     // publish; prev reads retired
        if (wave == 0 && jt + 1 < 128)
            async_copy_tile(vbase + (size_t)(jt + 1) * 2048, &vtile[cur ^ 1][0], lane);
        const _Float16* fp = &vtile[cur][lane * 16];
#else
        const _Float16* fp = vbase + (size_t)jt * 2048 + (size_t)lane * 16;
        if (jt + 1 < 128)
            __builtin_prefetch(vbase + (size_t)(jt + 1) * 2048 + (size_t)lane * 16, 0, 0);
#endif

        // Build A fragment: 16 softmax weights per lane, 16-bit A (16x32) layout.
        // half t -> K index: t + 2*hb
        v16h a;
        #pragma unroll
        for (int t = 0; t < 16; ++t) {
            int   kidx = t + 2 * hb;
            float d    = q - Ks[j0 + kidx];
            float s    = __expf(-d * d);          // RBF score in (0,1]
            float u    = s * 0.125f;              // in (0, 0.125]
            float w    = fmaf(u, fmaf(u, fmaf(u, 0.16666667f, 0.5f), 1.0f), 1.0f);
            rsp[t & 3] += w;
            a[t] = (_Float16)w;
        }

        v16h b0 = *(const v16h*)(fp);
        v16h b1 = *(const v16h*)(fp + 512);
        v16h b2 = *(const v16h*)(fp + 1024);
        v16h b3 = *(const v16h*)(fp + 1536);

        acc0 = __builtin_amdgcn_wmma_f32_16x16x32_f16(false, a, false, b0, (short)0, acc0, false, false);
        acc1 = __builtin_amdgcn_wmma_f32_16x16x32_f16(false, a, false, b1, (short)0, acc1, false, false);
        acc2 = __builtin_amdgcn_wmma_f32_16x16x32_f16(false, a, false, b2, (short)0, acc2, false, false);
        acc3 = __builtin_amdgcn_wmma_f32_16x16x32_f16(false, a, false, b3, (short)0, acc3, false, false);
    }

    // Row sums: lanes L and L^16 both hold partials for row L%16
    float rowsum = (rsp[0] + rsp[1]) + (rsp[2] + rsp[3]);
    float total  = rowsum + __shfl_xor(rowsum, 16);

    // Normalize and write PV in natural (B,N,DK) layout.
    const int col = lane & 15;
    #pragma unroll
    for (int r = 0; r < 8; ++r) {
        int   row = (lane < 16) ? r : r + 8;
        float inv = 1.0f / __shfl(total, row);
        size_t base = (((size_t)b * NTOK + i0 + row) << 6) + col;
        PV[base +  0] = acc0[r] * inv;
        PV[base + 16] = acc1[r] * inv;
        PV[base + 32] = acc2[r] * inv;
        PV[base + 48] = acc3[r] * inv;
    }
}

// ---------------------------------------------------------------------------
// Kernel 3: out = LayerNorm(PV + x) * gamma + beta   (one wave per 64-wide row)
// ---------------------------------------------------------------------------
__global__ void ln_kernel(const float* __restrict__ PV,
                          const float* __restrict__ x,
                          const float* __restrict__ gamma,
                          const float* __restrict__ beta,
                          float* __restrict__ out) {
    const int tid  = threadIdx.x;
    const int wave = tid >> 5;
    const int lane = tid & 31;
    const size_t row = (size_t)blockIdx.x * 4 + wave;

    const float* pr = PV + (row << 6);
    const float* xr = x  + (row << 6);
    float h0 = pr[lane]      + xr[lane];
    float h1 = pr[lane + 32] + xr[lane + 32];

    float s  = h0 + h1;
    float ss = h0 * h0 + h1 * h1;
    #pragma unroll
    for (int o = 16; o > 0; o >>= 1) {
        s  += __shfl_xor(s, o);
        ss += __shfl_xor(ss, o);
    }
    float mu  = s * (1.0f / 64.0f);
    float var = ss * (1.0f / 64.0f) - mu * mu;
    float rs  = rsqrtf(var + 1e-5f);

    out[(row << 6) + lane]      = (h0 - mu) * rs * gamma[lane]      + beta[lane];
    out[(row << 6) + lane + 32] = (h1 - mu) * rs * gamma[lane + 32] + beta[lane + 32];
}

// ---------------------------------------------------------------------------
extern "C" void kernel_launch(void* const* d_in, const int* in_sizes, int n_in,
                              void* d_out, int out_size, void* d_ws, size_t ws_size,
                              hipStream_t stream) {
    const float* x     = (const float*)d_in[0];
    const float* Wv    = (const float*)d_in[1];
    const float* bv    = (const float*)d_in[2];
    const float* wq    = (const float*)d_in[3];
    const float* wk    = (const float*)d_in[4];
    const float* gamma = (const float*)d_in[5];
    const float* beta  = (const float*)d_in[6];
    float* out = (float*)d_out;

    char* ws = (char*)d_ws;
    _Float16* Vfrag = (_Float16*)ws;                                   // 2 MB
    float*    Qb    = (float*)(ws + (size_t)2 * 1024 * 1024);          // 64 KB
    float*    Kb    = (float*)(ws + (size_t)2 * 1024 * 1024 + 65536);  // 64 KB
    float*    PV    = (float*)(ws + (size_t)2 * 1024 * 1024 + 131072); // 4 MB

    precompute_kernel<<<BATCH * (NTOK / 32), 256, 0, stream>>>(x, Wv, bv, wq, wk, Vfrag, Qb, Kb);
    attn_kernel<<<BATCH * (NTOK / 64), 128, 0, stream>>>(Vfrag, Qb, Kb, PV);
    ln_kernel<<<(BATCH * NTOK) / 4, 128, 0, stream>>>(PV, x, gamma, beta, out);
}
